// EuclideanGraphEncoder_28114855919656
// MI455X (gfx1250) — matrix operand order repaired and checked by reading, not compiled, not run on hardware
//
#include <hip/hip_runtime.h>

// CDNA5 / gfx1250: wave32, WMMA 16x16x32 f16 -> f32 accumulate.
typedef _Float16 v16h __attribute__((ext_vector_type(16)));
typedef _Float16 v8h  __attribute__((ext_vector_type(8)));
typedef float    v8f  __attribute__((ext_vector_type(8)));
typedef int      v4i  __attribute__((vector_size(16)));   // matches builtin param

#define GLOBAL_AS __attribute__((address_space(1)))
#define LDS_AS    __attribute__((address_space(3)))

#define GB   32     // batch
#define GN   1024   // nodes
#define GIN  64     // IN_DIM
#define GHID 128    // HID
#define GOUT 64     // 2*LAT

#if defined(__has_builtin)
#  if __has_builtin(__builtin_amdgcn_global_load_async_to_lds_b128) && \
      __has_builtin(__builtin_amdgcn_s_wait_asynccnt)
#    define USE_ASYNC_LDS 1
#  endif
#endif
#ifndef USE_ASYNC_LDS
#  define USE_ASYNC_LDS 0
#endif

// src[K][N] f32 -> dst[N][K] f16  (weights are tiny; done once per call)
__global__ void cvt_transpose_f16(const float* __restrict__ src,
                                  _Float16* __restrict__ dst, int K, int N) {
  int i = blockIdx.x * blockDim.x + threadIdx.x;
  if (i < K * N) {
    int k = i / N, n = i % N;
    dst[(size_t)n * K + k] = (_Float16)src[i];
  }
}

// C[M, N2T] = act( A[M, K] * B[K, N2T] + bias ), batch via blockIdx.y.
// A: f32 (on-the-fly cvt) or f16, row-major, lda = K.
// B: given TRANSPOSED in global: BT[N2T rows][K], row stride ldbT.
// Block tile 128 x N2T; 8 waves, wave = 16 rows x N2T cols.
// BT tile staged to LDS via async DMA (ASYNCcnt), double-buffered; K loop is
// unrolled by 2 so the buffer index and all pipeline guards are compile-time.
template <bool A_F32, bool HAS_BIAS, bool RELU, bool MASK, bool OUT_F32,
          bool OUT_T, int N2T>
__global__ __launch_bounds__(256) void gemm_wmma(
    const void* __restrict__ Abase, size_t aStride,
    const _Float16* __restrict__ BTbase, size_t bStride, int ldbT,
    const float* __restrict__ bias, const float* __restrict__ maskp,
    void* __restrict__ Cbase, size_t cStride, int ldcT, int K) {
  constexpr int NCT = N2T / 16;
  constexpr int BTS = 40;  // padded LDS row stride (halves); mult of 8 -> 16B rows
  constexpr int NBUF = USE_ASYNC_LDS ? 2 : 1;
  __shared__ _Float16 BT[NBUF][N2T * BTS];

  const int tid  = threadIdx.x;
  const int lane = tid & 31;
  const int wv   = tid >> 5;
  const int bidx = blockIdx.y;
  const int waveRow0 = blockIdx.x * 128 + wv * 16;
  const int m  = lane & 15;  // A row in wave tile / B col in col-tile
  const int hh = lane >> 4;  // K sub-block select per ISA fragment layout

  const _Float16* BbT = BTbase + (size_t)bidx * bStride;
  const size_t arow   = (size_t)(waveRow0 + m) * (size_t)K;
  const float*    A32 = (const float*)Abase + (size_t)bidx * aStride + arow;
  const _Float16* A16 = (const _Float16*)Abase + (size_t)bidx * aStride + arow;

  // stage BT[n=0..N2T, k0..k0+32) into LDS buffer: 16B contiguous chunks
  auto stage = [&](int buf, int k0) {
    constexpr int NCHUNK = (N2T * 4) / 256;  // exact: 2 for N2T=128, 1 for 64
#pragma unroll
    for (int i = 0; i < NCHUNK; ++i) {
      const int c    = tid + i * 256;
      const int n    = c >> 2;
      const int koff = (c & 3) << 3;
      const _Float16* g = BbT + (size_t)n * ldbT + (k0 + koff);
      _Float16* l = &BT[buf][n * BTS + koff];
#if USE_ASYNC_LDS
      __builtin_amdgcn_global_load_async_to_lds_b128(
          (GLOBAL_AS v4i*)g, (LDS_AS v4i*)l, 0, 0);
#else
      *(v8h*)l = *(const v8h*)g;
#endif
    }
  };

  // A fragment (16x32 f16), ISA layout: lane(m,hh):
  // halfs 0..7 = K(k0+hh*8 ..), halfs 8..15 = K(k0+16+hh*8 ..)
  auto loadA = [&](int k0) -> v16h {
    v16h a;
    if constexpr (A_F32) {
      const float* ap = A32 + k0 + hh * 8;
      const float4 f0 = *(const float4*)(ap);
      const float4 f1 = *(const float4*)(ap + 4);
      const float4 f2 = *(const float4*)(ap + 16);
      const float4 f3 = *(const float4*)(ap + 20);
      a[0]  = (_Float16)f0.x; a[1]  = (_Float16)f0.y;
      a[2]  = (_Float16)f0.z; a[3]  = (_Float16)f0.w;
      a[4]  = (_Float16)f1.x; a[5]  = (_Float16)f1.y;
      a[6]  = (_Float16)f1.z; a[7]  = (_Float16)f1.w;
      a[8]  = (_Float16)f2.x; a[9]  = (_Float16)f2.y;
      a[10] = (_Float16)f2.z; a[11] = (_Float16)f2.w;
      a[12] = (_Float16)f3.x; a[13] = (_Float16)f3.y;
      a[14] = (_Float16)f3.z; a[15] = (_Float16)f3.w;
    } else {
      const _Float16* ap = A16 + k0 + hh * 8;
      v8h lo = *(const v8h*)ap;
      v8h hi = *(const v8h*)(ap + 16);
      a = __builtin_shufflevector(lo, hi, 0, 1, 2, 3, 4, 5, 6, 7, 8, 9, 10, 11,
                                  12, 13, 14, 15);
    }
    return a;
  };

  v8f acc[NCT];
  const v8f zero = {0.f, 0.f, 0.f, 0.f, 0.f, 0.f, 0.f, 0.f};
#pragma unroll
  for (int ct = 0; ct < NCT; ++ct) acc[ct] = zero;

  const int nsteps = K >> 5;  // 2 (embed), 4 (msg/proj), 32 (agg): always even
#if USE_ASYNC_LDS
  stage(0, 0);  // prologue DMA
#endif
  v16h acur = loadA(0);

  // one pipeline step; cb and hasNext are literals at every call site
  auto step = [&](int s, int cb, bool hasNext) {
#if USE_ASYNC_LDS
    __builtin_amdgcn_s_wait_asynccnt(0);  // own DMA for buf[cb] done
    __syncthreads();                      // everyone's DMA visible; prev reads done
    if (hasNext) stage(cb ^ 1, (s + 1) << 5);  // overlap next DMA with compute
#else
    cb = 0;
    __syncthreads();
    stage(0, s << 5);
    __syncthreads();
#endif
    // prefetch next A fragment while we chew on LDS + WMMA
    v16h anext = acur;
    if (hasNext) anext = loadA((s + 1) << 5);

    v16h bf[NCT];
#pragma unroll
    for (int ct = 0; ct < NCT; ++ct) {
      const _Float16* bp = &BT[cb][(ct * 16 + m) * BTS + (hh << 3)];
      v8h b0 = *(const v8h*)bp;
      v8h b1 = *(const v8h*)(bp + 16);
      bf[ct] = __builtin_shufflevector(b0, b1, 0, 1, 2, 3, 4, 5, 6, 7, 8, 9, 10,
                                       11, 12, 13, 14, 15);
    }
#pragma unroll
    for (int ct = 0; ct < NCT; ++ct)
      acc[ct] = __builtin_amdgcn_wmma_f32_16x16x32_f16(
          false, acur, false, bf[ct], (short)0, acc[ct], false, false);
    acur = anext;
  };

  int s = 0;
  for (; s + 2 < nsteps; s += 2) {  // body pairs, always prefetching
    step(s, 0, true);
    step(s + 1, 1, true);
  }
  step(s, 0, true);        // stages + prefetches the final step
  step(s + 1, 1, false);   // final step: no further prefetch

  // epilogue: C/D layout -> VGPR r holds (M = waveRow0 + r + 8*hh, N = m)
  const int rOff = hh << 3;
  if constexpr (OUT_T) {
    // transposed f16 output: CT[col][row]; 8 rows per lane contiguous -> b128 store
    _Float16* CT = (_Float16*)Cbase + (size_t)bidx * cStride;
#pragma unroll
    for (int ct = 0; ct < NCT; ++ct) {
      const int col = ct * 16 + m;
      const float bv = HAS_BIAS ? bias[col] : 0.0f;
      v8h pk;
#pragma unroll
      for (int r = 0; r < 8; ++r) {
        float v = acc[ct][r] + bv;
        if (RELU) v = fmaxf(v, 0.0f);
        pk[r] = (_Float16)v;
      }
      *(v8h*)&CT[(size_t)col * ldcT + (waveRow0 + rOff)] = pk;
    }
  } else if constexpr (OUT_F32) {
    float* C = (float*)Cbase + (size_t)bidx * cStride;
#pragma unroll
    for (int ct = 0; ct < NCT; ++ct) {
      const int col = ct * 16 + m;
      const float bv = HAS_BIAS ? bias[col] : 0.0f;
#pragma unroll
      for (int r = 0; r < 8; ++r) {
        const int row = waveRow0 + rOff + r;
        float v = acc[ct][r] + bv;
        if (RELU) v = fmaxf(v, 0.0f);
        if (MASK) v *= maskp[row];
        C[(size_t)row * N2T + col] = v;
      }
    }
  } else {
    _Float16* C = (_Float16*)Cbase + (size_t)bidx * cStride;
#pragma unroll
    for (int ct = 0; ct < NCT; ++ct) {
      const int col = ct * 16 + m;
      const float bv = HAS_BIAS ? bias[col] : 0.0f;
#pragma unroll
      for (int r = 0; r < 8; ++r) {
        const int row = waveRow0 + rOff + r;
        float v = acc[ct][r] + bv;
        if (RELU) v = fmaxf(v, 0.0f);
        C[(size_t)row * N2T + col] = (_Float16)v;
      }
    }
  }
}

extern "C" void kernel_launch(void* const* d_in, const int* in_sizes, int n_in,
                              void* d_out, int out_size, void* d_ws,
                              size_t ws_size, hipStream_t stream) {
  const float* x       = (const float*)d_in[0];  // [B,N,64]
  const float* adj     = (const float*)d_in[1];  // [B,N,N]
  const float* mask    = (const float*)d_in[2];  // [B,N,1]
  const float* W_embed = (const float*)d_in[3];  // [64,128]
  const float* Wl      = (const float*)d_in[4];  // [3,128,128]
  const float* bl      = (const float*)d_in[5];  // [3,128]
  const float* W_proj  = (const float*)d_in[6];  // [128,64]
  const float* b_proj  = (const float*)d_in[7];  // [64]

  // workspace carve-up (~17 MiB)
  char* ws = (char*)d_ws;
  _Float16* wEmbT  = (_Float16*)ws; ws += (((size_t)GIN * GHID * 2) + 255) & ~(size_t)255;
  _Float16* wlT    = (_Float16*)ws; ws += (((size_t)3 * GHID * GHID * 2) + 255) & ~(size_t)255;
  _Float16* wProjT = (_Float16*)ws; ws += (((size_t)GHID * GOUT * 2) + 255) & ~(size_t)255;
  _Float16* h16    = (_Float16*)ws; ws += (size_t)GB * GN * GHID * 2;   // [B*N][HID]
  _Float16* msgT   = (_Float16*)ws;                                     // [HID][B*N]

  const int Mflat = GB * GN;  // 32768
  const dim3 blk(256);

  // 1) weights f32 -> f16, transposed to [N][K]
  cvt_transpose_f16<<<(GIN * GHID + 255) / 256, 256, 0, stream>>>(W_embed, wEmbT, GIN, GHID);
  for (int l = 0; l < 3; ++l)
    cvt_transpose_f16<<<(GHID * GHID + 255) / 256, 256, 0, stream>>>(
        Wl + (size_t)l * GHID * GHID, wlT + (size_t)l * GHID * GHID, GHID, GHID);
  cvt_transpose_f16<<<(GHID * GOUT + 255) / 256, 256, 0, stream>>>(W_proj, wProjT, GHID, GOUT);

  // 2) embed: h = x @ W_embed   (A f32; BT = W_embed^T [128][64])
  gemm_wmma<true, false, false, false, false, false, GHID>
      <<<dim3(Mflat / 128, 1), blk, 0, stream>>>(
          x, 0, wEmbT, 0, GIN, nullptr, nullptr, h16, 0, 0, GIN);

  // 3) GCN layers
  for (int l = 0; l < 3; ++l) {
    // msg^T = (h @ Wl[l] + bl[l])^T  -> msgT [HID][B*N] (vectorized b128 stores)
    gemm_wmma<false, true, false, false, false, true, GHID>
        <<<dim3(Mflat / 128, 1), blk, 0, stream>>>(
            h16, 0, wlT + (size_t)l * GHID * GHID, 0, GHID,
            bl + (size_t)l * GHID, nullptr, msgT, 0, Mflat, GHID);
    // h = relu(adj @ msg)  (batched; B operand = msgT rows, batch col-offset b*N)
    gemm_wmma<true, false, true, false, false, false, GHID>
        <<<dim3(GN / 128, GB), blk, 0, stream>>>(
            adj, (size_t)GN * GN, msgT, (size_t)GN, Mflat,
            nullptr, nullptr, h16, (size_t)GN * GHID, 0, GN);
  }

  // 4) out = (h @ W_proj + b_proj) * mask   (f32 out)
  gemm_wmma<false, true, false, true, true, false, GOUT>
      <<<dim3(Mflat / 128, 1), blk, 0, stream>>>(
          h16, 0, wProjT, 0, GHID, b_proj, mask, d_out, 0, 0, GHID);
}
